// SelectiveSSM_652835029108
// MI455X (gfx1250) — compile-verified, matrix operand order checked by reading, not run on hardware
//
#include <hip/hip_runtime.h>
#include <hip/hip_bf16.h>

// ---------------------------------------------------------------------------
// Mamba-style selective SSM forward for MI455X (gfx1250), wave32 + WMMA.
//   xz  = x @ W_in^T                      (bf16 WMMA, f32 accum)
//   xs  = silu(causal_dwconv(xi)+b)       (f32, bandwidth)
//   xdb = xs @ W_x^T                      (f32 wave-reduce, N=33)
//   scan over L (h[16] in VGPRs), fused D-skip + silu(z) gate -> g (bf16)
//   out = g @ W_out^T                     (bf16 WMMA, f32 accum)
// ---------------------------------------------------------------------------

#define BATCH   8
#define SEQ     2048
#define DMODEL  1024
#define DI      2048          // d_inner
#define NSTATE  16
#define MTOT    (BATCH * SEQ) // 16384 rows

typedef __bf16 bf16;
typedef __attribute__((ext_vector_type(16))) __bf16 v16bf;
typedef __attribute__((ext_vector_type(8)))  __bf16 v8bf;
typedef __attribute__((ext_vector_type(8)))  float  v8f;

// ---- fragment load: 16-bit A/B matrix layout (ISA 7.12.2) -----------------
// lane = row (A) / col (B) in [0,15]; lanes>=16 take K skewed by +8.
// halves 0..7  <- K = base+0..7 ; halves 8..15 <- K = base+16..23
static __device__ inline v16bf ldfrag(const bf16* __restrict__ p) {
    v8bf lo = *(const v8bf*)(p);
    v8bf hi = *(const v8bf*)(p + 16);
    return __builtin_shufflevector(lo, hi, 0, 1, 2, 3, 4, 5, 6, 7,
                                           8, 9, 10, 11, 12, 13, 14, 15);
}

#define WMMA_BF16(Acc, Af, Bf) \
    __builtin_amdgcn_wmma_f32_16x16x32_bf16(false, (Af), false, (Bf), \
                                            (short)0, (Acc), false, false)

// ---------------------------------------------------------------------------
// GEMM: C[M,N] (f32) = A[M,K] (bf16, row-major) * B[N,K]^T (bf16, row-major)
// block = 256 threads = 8 waves (2 in M x 4 in N); wave tile 32(M) x 64(N);
// block tile 64 x 256.  8 WMMAs per k-step against 6 fragment loads, with an
// explicit register double-buffer so next-k loads overlap current-k WMMAs.
// Requires M%64==0, N%256==0, K%32==0 (true for all uses here).
// ---------------------------------------------------------------------------
__global__ __launch_bounds__(256)
void ssm_gemm_bf16_wmma(const bf16* __restrict__ A, const bf16* __restrict__ Bm,
                        float* __restrict__ C, int M, int N, int K) {
    const int lane  = threadIdx.x & 31;
    const int wave  = threadIdx.x >> 5;
    const int waveM = wave & 1;   // 0..1
    const int waveN = wave >> 1;  // 0..3
    const int r     = lane & 15;
    const int hi    = lane >> 4;  // 0/1 -> K skew of 8

    const int mBase = blockIdx.y * 64  + waveM * 32;
    const int nBase = blockIdx.x * 256 + waveN * 64;

    const size_t rowK = (size_t)K;
    const bf16* aRow0 = A  + (size_t)(mBase + r) * rowK + hi * 8;
    const bf16* aRow1 = aRow0 + 16 * rowK;
    const bf16* bRow0 = Bm + (size_t)(nBase + r) * rowK + hi * 8;
    const bf16* bRow1 = bRow0 + 16 * rowK;
    const bf16* bRow2 = bRow0 + 32 * rowK;
    const bf16* bRow3 = bRow0 + 48 * rowK;

    v8f acc00 = {}, acc01 = {}, acc02 = {}, acc03 = {};
    v8f acc10 = {}, acc11 = {}, acc12 = {}, acc13 = {};

    // prologue: load k=0 fragments
    v16bf aC0 = ldfrag(aRow0), aC1 = ldfrag(aRow1);
    v16bf bC0 = ldfrag(bRow0), bC1 = ldfrag(bRow1);
    v16bf bC2 = ldfrag(bRow2), bC3 = ldfrag(bRow3);

    for (int k = 32; k < K; k += 32) {
        // speculative prefetch two k-steps ahead (dropped silently if OOB)
        __builtin_prefetch(aRow0 + k + 32, 0, 1);
        __builtin_prefetch(aRow1 + k + 32, 0, 1);
        __builtin_prefetch(bRow0 + k + 32, 0, 1);
        __builtin_prefetch(bRow2 + k + 32, 0, 1);

        // issue next-k loads before consuming current fragments
        v16bf aN0 = ldfrag(aRow0 + k), aN1 = ldfrag(aRow1 + k);
        v16bf bN0 = ldfrag(bRow0 + k), bN1 = ldfrag(bRow1 + k);
        v16bf bN2 = ldfrag(bRow2 + k), bN3 = ldfrag(bRow3 + k);

        acc00 = WMMA_BF16(acc00, aC0, bC0);
        acc01 = WMMA_BF16(acc01, aC0, bC1);
        acc02 = WMMA_BF16(acc02, aC0, bC2);
        acc03 = WMMA_BF16(acc03, aC0, bC3);
        acc10 = WMMA_BF16(acc10, aC1, bC0);
        acc11 = WMMA_BF16(acc11, aC1, bC1);
        acc12 = WMMA_BF16(acc12, aC1, bC2);
        acc13 = WMMA_BF16(acc13, aC1, bC3);

        aC0 = aN0; aC1 = aN1;
        bC0 = bN0; bC1 = bN1; bC2 = bN2; bC3 = bN3;
    }

    // epilogue k-step
    acc00 = WMMA_BF16(acc00, aC0, bC0);
    acc01 = WMMA_BF16(acc01, aC0, bC1);
    acc02 = WMMA_BF16(acc02, aC0, bC2);
    acc03 = WMMA_BF16(acc03, aC0, bC3);
    acc10 = WMMA_BF16(acc10, aC1, bC0);
    acc11 = WMMA_BF16(acc11, aC1, bC1);
    acc12 = WMMA_BF16(acc12, aC1, bC2);
    acc13 = WMMA_BF16(acc13, aC1, bC3);

    // C/D layout: VGPR i -> row = i + 8*hi, col = r (ISA 7.12.2)
    float* c0 = C + (size_t)(mBase + hi * 8) * N + nBase + r;
    float* c1 = c0 + (size_t)16 * N;   // second 16-row A tile
    #pragma unroll
    for (int i = 0; i < 8; ++i) {
        float* cr0 = c0 + (size_t)i * N;
        float* cr1 = c1 + (size_t)i * N;
        cr0[0]  = acc00[i];
        cr0[16] = acc01[i];
        cr0[32] = acc02[i];
        cr0[48] = acc03[i];
        cr1[0]  = acc10[i];
        cr1[16] = acc11[i];
        cr1[32] = acc12[i];
        cr1[48] = acc13[i];
    }
}

// ---------------------------------------------------------------------------
// float -> bf16 cast
// ---------------------------------------------------------------------------
__global__ __launch_bounds__(256)
void ssm_cast_bf16(const float* __restrict__ src, bf16* __restrict__ dst, size_t n) {
    size_t i = (size_t)blockIdx.x * blockDim.x + threadIdx.x;
    if (i < n) dst[i] = (bf16)src[i];
}

// ---------------------------------------------------------------------------
// causal depthwise conv (K=4) + bias + SiLU; xz layout [B, L, 2*DI], xi = low DI
// ---------------------------------------------------------------------------
__global__ __launch_bounds__(256)
void ssm_conv_silu(const float* __restrict__ xz, const float* __restrict__ cw,
                   const float* __restrict__ cb, float* __restrict__ xs) {
    size_t idx = (size_t)blockIdx.x * blockDim.x + threadIdx.x;  // over B*L*DI
    if (idx >= (size_t)MTOT * DI) return;
    int d     = (int)(idx % DI);
    size_t bl = idx / DI;                 // b*L + l
    int l     = (int)(bl % SEQ);

    float w0 = cw[d * 4 + 0], w1 = cw[d * 4 + 1];
    float w2 = cw[d * 4 + 2], w3 = cw[d * 4 + 3];
    const float* p = xz + bl * (size_t)(2 * DI) + d;

    float acc = cb[d] + w3 * p[0];
    if (l > 0) acc += w2 * p[-(ptrdiff_t)(2 * DI)];
    if (l > 1) acc += w1 * p[-(ptrdiff_t)(4 * DI)];
    if (l > 2) acc += w0 * p[-(ptrdiff_t)(6 * DI)];

    xs[idx] = acc / (1.f + __expf(-acc));   // silu
}

// ---------------------------------------------------------------------------
// x_dbl[m, 0..32] = xs[m,:] @ W_x[j,:]  (N=33 skinny GEMM, wave reductions)
// one block per row m; 8 waves; wave w covers j = w, w+8, ...
// ---------------------------------------------------------------------------
__global__ __launch_bounds__(256)
void ssm_xdbl(const float* __restrict__ xs, const float* __restrict__ Wx,
              float* __restrict__ xdbl) {
    const int m    = blockIdx.x;
    const int lane = threadIdx.x & 31;
    const int wave = threadIdx.x >> 5;
    const float* xrow = xs + (size_t)m * DI;
    for (int j = wave; j < 2 * NSTATE + 1; j += 8) {
        const float* wrow = Wx + (size_t)j * DI;
        float acc = 0.f;
        for (int k = lane; k < DI; k += 32)
            acc = fmaf(xrow[k], wrow[k], acc);
        #pragma unroll
        for (int off = 16; off > 0; off >>= 1)
            acc += __shfl_down(acc, off, 32);
        if (lane == 0) xdbl[(size_t)m * 33 + j] = acc;
    }
}

// ---------------------------------------------------------------------------
// Selective scan, fused with dt=softplus(...), D-skip, silu(z) gate, bf16 cast.
// grid = 16 blocks (b in 0..7, half of DI); block = 1024 threads (one channel each).
// h[16] kept in VGPRs across all 2048 steps.
// ---------------------------------------------------------------------------
__global__ __launch_bounds__(1024)
void ssm_scan_gate(const float* __restrict__ xdbl, const float* __restrict__ xs,
                   const float* __restrict__ xz,   const float* __restrict__ Wdt,
                   const float* __restrict__ bdt,  const float* __restrict__ Alog,
                   const float* __restrict__ Dp,   bf16* __restrict__ g) {
    const int b = blockIdx.x >> 1;
    const int d = (blockIdx.x & 1) * 1024 + threadIdx.x;

    float A[NSTATE], h[NSTATE];
    #pragma unroll
    for (int n = 0; n < NSTATE; ++n) {
        A[n] = -__expf(Alog[(size_t)d * NSTATE + n]);
        h[n] = 0.f;
    }
    const float wdt = Wdt[d], bd = bdt[d], Dd = Dp[d];

    for (int l = 0; l < SEQ; ++l) {
        const size_t bl = (size_t)b * SEQ + l;
        const float* xd = xdbl + bl * 33;       // block-uniform address

        float dtp = fmaf(xd[0], wdt, bd);
        float dt  = (dtp > 20.f) ? dtp : log1pf(__expf(dtp));   // softplus
        float xv  = xs[bl * DI + d];
        float dx  = dt * xv;

        float y = 0.f;
        #pragma unroll
        for (int n = 0; n < NSTATE; ++n) {
            float ab = __expf(dt * A[n]);
            h[n] = fmaf(ab, h[n], dx * xd[1 + n]);        // B_t
            y    = fmaf(h[n], xd[17 + n], y);             // C_t
        }
        y = fmaf(Dd, xv, y);

        float zv   = xz[bl * (size_t)(2 * DI) + DI + d];
        float gate = zv / (1.f + __expf(-zv));            // silu(z)
        g[bl * DI + d] = (bf16)(y * gate);
    }
}

// ---------------------------------------------------------------------------
// host launcher
// ---------------------------------------------------------------------------
extern "C" void kernel_launch(void* const* d_in, const int* in_sizes, int n_in,
                              void* d_out, int out_size, void* d_ws, size_t ws_size,
                              hipStream_t stream) {
    (void)in_sizes; (void)n_in; (void)out_size; (void)ws_size;
    const float* x     = (const float*)d_in[0];  // [B,L,DMODEL]
    const float* W_in  = (const float*)d_in[1];  // [2*DI, DMODEL]
    const float* convw = (const float*)d_in[2];  // [DI,1,4]
    const float* convb = (const float*)d_in[3];  // [DI]
    const float* W_x   = (const float*)d_in[4];  // [33, DI]
    const float* W_dt  = (const float*)d_in[5];  // [DI,1]
    const float* b_dt  = (const float*)d_in[6];  // [DI]
    const float* A_log = (const float*)d_in[7];  // [DI,16]
    const float* Dp    = (const float*)d_in[8];  // [DI]
    const float* W_out = (const float*)d_in[9];  // [DMODEL, DI]
    float* out = (float*)d_out;                  // [B,L,DMODEL]

    char* ws = (char*)d_ws;
    size_t off = 0;
    auto carve = [&](size_t bytes) -> void* {
        void* p = ws + off;
        off += (bytes + 255) & ~(size_t)255;
        return p;
    };
    bf16*  xbf   = (bf16*)carve((size_t)MTOT * DMODEL * sizeof(bf16));      // 33.5 MB
    bf16*  wibf  = (bf16*)carve((size_t)2 * DI * DMODEL * sizeof(bf16));    //  8.4 MB
    bf16*  wobf  = (bf16*)carve((size_t)DMODEL * DI * sizeof(bf16));        //  4.2 MB
    float* xzf   = (float*)carve((size_t)MTOT * 2 * DI * sizeof(float));    // 268 MB
    float* xsf   = (float*)carve((size_t)MTOT * DI * sizeof(float));        // 134 MB
    float* xdblf = (float*)carve((size_t)MTOT * 33 * sizeof(float));        //  2.2 MB
    bf16*  gbf   = (bf16*)carve((size_t)MTOT * DI * sizeof(bf16));          // 67.1 MB

    const int B256 = 256;

    // 1) casts to bf16
    {
        size_t n = (size_t)MTOT * DMODEL;
        ssm_cast_bf16<<<(unsigned)((n + B256 - 1) / B256), B256, 0, stream>>>(x, xbf, n);
        n = (size_t)2 * DI * DMODEL;
        ssm_cast_bf16<<<(unsigned)((n + B256 - 1) / B256), B256, 0, stream>>>(W_in, wibf, n);
        n = (size_t)DMODEL * DI;
        ssm_cast_bf16<<<(unsigned)((n + B256 - 1) / B256), B256, 0, stream>>>(W_out, wobf, n);
    }

    // 2) xz = x @ W_in^T : M=16384, N=4096, K=1024
    {
        dim3 grid(2 * DI / 256, MTOT / 64);
        ssm_gemm_bf16_wmma<<<grid, B256, 0, stream>>>(xbf, wibf, xzf,
                                                      MTOT, 2 * DI, DMODEL);
    }

    // 3) xs = silu(conv(xi) + b)
    {
        size_t n = (size_t)MTOT * DI;
        ssm_conv_silu<<<(unsigned)((n + B256 - 1) / B256), B256, 0, stream>>>(
            xzf, convw, convb, xsf);
    }

    // 4) x_dbl = xs @ W_x^T (N=33)
    ssm_xdbl<<<MTOT, B256, 0, stream>>>(xsf, W_x, xdblf);

    // 5) selective scan + gate -> g (bf16)
    ssm_scan_gate<<<BATCH * 2, 1024, 0, stream>>>(xdblf, xsf, xzf, W_dt, b_dt,
                                                  A_log, Dp, gbf);

    // 6) out = g @ W_out^T : M=16384, N=1024, K=2048
    {
        dim3 grid(DMODEL / 256, MTOT / 64);
        ssm_gemm_bf16_wmma<<<grid, B256, 0, stream>>>(gbf, wobf, out,
                                                      MTOT, DMODEL, DI);
    }
}
